// Lstm_13975823581927
// MI455X (gfx1250) — compile-verified, matrix operand order checked by reading, not run on hardware
//
#include <hip/hip_runtime.h>

// ---------------------------------------------------------------------------
// LSTM sentence classifier for MI455X (gfx1250, wave32, WMMA f16 16x16x32).
//
//  1) wswizzle:        W_lstm (600x1200 f32) -> Wx,Wh f16 pre-swizzled into the
//                      CDNA5 WMMA B-fragment per-lane layout (K padded 300->320).
//  2) embed_gemm:      xW[t][b][n] = emb[ids[b,t]] @ Wx + b_lstm (f16 out).
//                      Embedding rows staged global->LDS with CDNA5 async-to-LDS.
//  3) lstm_persistent: ONE kernel, 8 blocks x 16 batch rows. Batch rows are
//                      independent in the recurrence, so each block runs all
//                      512 timesteps with h resident in LDS (no global sync).
//  4) final_dense:     300x3 dense + log_softmax + mean NLL.
// ---------------------------------------------------------------------------

typedef __attribute__((ext_vector_type(16))) _Float16 v16h;
typedef __attribute__((ext_vector_type(8)))  _Float16 v8h;
typedef __attribute__((ext_vector_type(8)))  float    v8f;
typedef __attribute__((ext_vector_type(4)))  int      v4i;

#define B_   128
#define T_   512
#define D_   300
#define H_   300
#define NG   1200       // 4*H
#define KP   320        // K padded to multiple of 32
#define KT   10         // K tiles (32 each)
#define NT   75         // N tiles (16 each)
#define FORGET_BIAS 1.0f

#if defined(__HIP_DEVICE_COMPILE__) && __has_builtin(__builtin_amdgcn_global_load_async_to_lds_b128)
#define HAS_ASYNC_LDS 1
typedef __attribute__((address_space(1))) v4i* as1_v4i;
typedef __attribute__((address_space(3))) v4i* as3_v4i;
__device__ __forceinline__ void async_ldg_b128(const void* g, const void* l) {
    // Global pointer value preserved through a 64-bit integer; the LDS byte
    // offset is addr[31:0] of the generic address per the ISA flat-aperture
    // mapping, so truncation to a 32-bit as(3) pointer is correct.
    __builtin_amdgcn_global_load_async_to_lds_b128(
        (as1_v4i)(unsigned long long)(uintptr_t)g,
        (as3_v4i)(unsigned int)(uintptr_t)l, 0, 0);
}
__device__ __forceinline__ void wait_async0() {
#if __has_builtin(__builtin_amdgcn_s_wait_asynccnt)
    __builtin_amdgcn_s_wait_asynccnt(0);
#else
    asm volatile("s_wait_asynccnt 0" ::: "memory");
#endif
}
#endif

// k index inside a 32-wide K tile for fragment element e of `lane`
// (per CDNA5 ISA 16-bit A/B matrix VGPR layout tables).
__device__ __forceinline__ int frag_k(int lane, int e) {
    return (e & 7) + ((e >> 3) << 4) + ((lane >> 4) << 3);
}

// --------------------------------------------------------------------------
// Pre-swizzle W_lstm into WMMA B-fragment layout, f16, zero-padded K.
// Frag index: ((kt*NT + nt)*32 + lane)*16 + e   (16 contiguous f16 per lane).
// --------------------------------------------------------------------------
__global__ void __launch_bounds__(256)
wswizzle(const float* __restrict__ W, _Float16* __restrict__ Wx,
         _Float16* __restrict__ Wh) {
    int idx = blockIdx.x * blockDim.x + threadIdx.x;
    if (idx >= KT * NT * 32 * 16) return;
    int e    = idx & 15;
    int lane = (idx >> 4) & 31;
    int nt   = (idx >> 9) % NT;
    int kt   = (idx >> 9) / NT;
    int k    = kt * 32 + frag_k(lane, e);
    int n    = nt * 16 + (lane & 15);
    float vx = (k < D_) ? W[(size_t)k * NG + n] : 0.0f;          // rows 0..299
    float vh = (k < H_) ? W[(size_t)(D_ + k) * NG + n] : 0.0f;   // rows 300..599
    Wx[idx] = (_Float16)vx;
    Wh[idx] = (_Float16)vh;
}

// --------------------------------------------------------------------------
// xW[t][b][:] = emb[ids[b,t]] @ Wx + b_lstm  (f16 output).
// One block = 16 rows (same t, 16 consecutive b), 4 waves split the N tiles.
// --------------------------------------------------------------------------
__global__ void __launch_bounds__(128)
embed_gemm(const int* __restrict__ ids, const float* __restrict__ emb,
           const _Float16* __restrict__ WxSw, const float* __restrict__ b_lstm,
           _Float16* __restrict__ xw) {
    __shared__ __align__(16) _Float16 xs[16 * KP];      // 10.0 KB
#ifdef HAS_ASYNC_LDS
    __shared__ __align__(16) float    stage[16 * D_];   // 18.75 KB
#endif

    int tile = blockIdx.x;           // 0 .. 4095
    int t    = tile >> 3;            // 8 row-tiles per timestep (B=128)
    int brow = (tile & 7) << 4;
    int tid  = threadIdx.x;

#ifdef HAS_ASYNC_LDS
    // Async-copy 16 embedding rows (1200 B each = 75 x b128) into LDS.
    for (int i = tid; i < 16 * 75; i += 128) {
        int r = i / 75, ch = i % 75;
        int id = ids[(size_t)(brow + r) * T_ + t];
        async_ldg_b128(emb + (size_t)id * D_ + ch * 4, &stage[r * D_ + ch * 4]);
    }
    wait_async0();
    __syncthreads();
    for (int i = tid; i < 16 * KP; i += 128) {
        int r = i / KP, k = i % KP;
        xs[i] = (_Float16)((k < D_) ? stage[r * D_ + k] : 0.0f);
    }
#else
    for (int i = tid; i < 16 * KP; i += 128) {
        int r = i / KP, k = i % KP;
        float v = 0.0f;
        if (k < D_) {
            int id = ids[(size_t)(brow + r) * T_ + t];
            v = emb[(size_t)id * D_ + k];
        }
        xs[i] = (_Float16)v;
    }
#endif
    __syncthreads();

    int lane = tid & 31, wave = tid >> 5;
    int row  = lane & 15;
    int klo  = (lane >> 4) << 3;     // 0 or 8

    for (int nt = wave; nt < NT; nt += 4) {
        v8f acc = {};
        const _Float16* bp = &WxSw[((size_t)nt * 32 + lane) * 16];
        #pragma unroll
        for (int kt = 0; kt < KT; ++kt) {
            const _Float16* ap = &xs[row * KP + kt * 32 + klo];
            v8h alo = *(const v8h*)ap;
            v8h ahi = *(const v8h*)(ap + 16);
            v16h a;
            #pragma unroll
            for (int i = 0; i < 8; ++i) { a[i] = alo[i]; a[i + 8] = ahi[i]; }
            v16h b = *(const v16h*)(bp + kt * (NT * 512));   // immediate offsets
            acc = __builtin_amdgcn_wmma_f32_16x16x32_f16(
                      false, a, false, b, (short)0, acc, false, false);
        }
        int   n    = nt * 16 + (lane & 15);
        float bias = b_lstm[n];
        #pragma unroll
        for (int r8 = 0; r8 < 8; ++r8) {
            int m = r8 + ((lane >> 4) << 3);   // C/D layout: M = vgpr + 8*(lane/16)
            xw[((size_t)t * B_ + brow + m) * NG + n] = (_Float16)(acc[r8] + bias);
        }
    }
}

// --------------------------------------------------------------------------
// Persistent recurrence: each block owns 16 independent batch rows and runs
// all T_ steps with h in LDS. z = h @ Wh + xW[t]; gates; masked h/c update.
// --------------------------------------------------------------------------
__global__ void __launch_bounds__(256)
lstm_persistent(const _Float16* __restrict__ WhSw, const _Float16* __restrict__ xw,
                const int* __restrict__ lengths, _Float16* __restrict__ hout,
                float* __restrict__ c) {
    __shared__ __align__(16) _Float16 hs[16 * KP];   // 10.0 KB, persistent h
    __shared__ __align__(16) _Float16 zs[16 * NG];   // 37.5 KB, per-step z
    __shared__ int len16[16];

    int brow = blockIdx.x << 4;
    int tid  = threadIdx.x;
    int lane = tid & 31, wave = tid >> 5;
    int row  = lane & 15;
    int klo  = (lane >> 4) << 3;

    // Init state: h = 0 (incl. K pad), c = 0, cache lengths.
    for (int i = tid; i < 16 * KP; i += 256) hs[i] = (_Float16)0.0f;
    for (int i = tid; i < 16 * H_; i += 256)
        c[(size_t)(brow + i / H_) * H_ + (i % H_)] = 0.0f;
    if (tid < 16) len16[tid] = lengths[brow + tid];
    __syncthreads();

    for (int t = 0; t < T_; ++t) {
        const _Float16* xwt = xw + ((size_t)t * B_ + brow) * NG;
        if (t + 1 < T_)  // pull next step's 38.4 KB xW slab toward L2
            __builtin_prefetch(xw + ((size_t)(t + 1) * B_ + brow) * NG + tid * 75, 0, 1);

        // z tiles: 75 N-tiles split over 8 waves, K = 10 x 32 (f16 WMMA).
        for (int nt = wave; nt < NT; nt += 8) {
            v8f acc = {};
            const _Float16* bp = &WhSw[((size_t)nt * 32 + lane) * 16];
            #pragma unroll
            for (int kt = 0; kt < KT; ++kt) {
                const _Float16* ap = &hs[row * KP + kt * 32 + klo];
                v8h alo = *(const v8h*)ap;
                v8h ahi = *(const v8h*)(ap + 16);
                v16h a;
                #pragma unroll
                for (int i = 0; i < 8; ++i) { a[i] = alo[i]; a[i + 8] = ahi[i]; }
                v16h b = *(const v16h*)(bp + kt * (NT * 512));
                acc = __builtin_amdgcn_wmma_f32_16x16x32_f16(
                          false, a, false, b, (short)0, acc, false, false);
            }
            int n = nt * 16 + (lane & 15);
            #pragma unroll
            for (int r8 = 0; r8 < 8; ++r8) {
                int m = r8 + ((lane >> 4) << 3);
                zs[m * NG + n] = (_Float16)(acc[r8] + (float)xwt[(size_t)m * NG + n]);
            }
        }
        __syncthreads();   // z complete; also protects hs reads above

        // Gates: i | j | f | o at u, u+300, u+600, u+900.
        for (int i = tid; i < 16 * H_; i += 256) {
            int m = i / H_, u = i % H_;
            float iz = (float)zs[m * NG + u];
            float jz = (float)zs[m * NG + H_ + u];
            float fz = (float)zs[m * NG + 2 * H_ + u];
            float oz = (float)zs[m * NG + 3 * H_ + u];
            size_t ci = (size_t)(brow + m) * H_ + u;
            float cold = c[ci];
            float sf = 1.0f / (1.0f + __expf(-(fz + FORGET_BIAS)));
            float si = 1.0f / (1.0f + __expf(-iz));
            float so = 1.0f / (1.0f + __expf(-oz));
            float cn = cold * sf + si * tanhf(jz);
            float hn = tanhf(cn) * so;
            if (t < len16[m]) {
                c[ci] = cn;
                hs[m * KP + u] = (_Float16)hn;
            }
        }
        __syncthreads();   // hs/zs settled before next step's GEMM
    }

    // Export final h (pad cols still zero).
    for (int i = tid; i < 16 * KP; i += 256)
        hout[(size_t)(brow + i / KP) * KP + (i % KP)] = hs[i];
}

// --------------------------------------------------------------------------
// logits = h @ W_dense + b_dense; log_softmax; mean NLL. C=3 -> scalar code.
// --------------------------------------------------------------------------
__global__ void __launch_bounds__(128)
final_dense(const _Float16* __restrict__ h, const float* __restrict__ Wd,
            const float* __restrict__ bd, const int* __restrict__ labels,
            float* __restrict__ out) {
    __shared__ float red[128];
    int b = threadIdx.x;
    float lg[3] = { bd[0], bd[1], bd[2] };
    for (int u = 0; u < H_; ++u) {
        float hv = (float)h[(size_t)b * KP + u];
        #pragma unroll
        for (int cc = 0; cc < 3; ++cc) lg[cc] += hv * Wd[u * 3 + cc];
    }
    float mx  = fmaxf(lg[0], fmaxf(lg[1], lg[2]));
    float se  = __expf(lg[0] - mx) + __expf(lg[1] - mx) + __expf(lg[2] - mx);
    float lse = mx + __logf(se);
    int lab   = labels[b];
    red[b]    = lse - lg[lab];                   // nll
    #pragma unroll
    for (int cc = 0; cc < 3; ++cc) out[1 + b * 3 + cc] = lg[cc];
    __syncthreads();
    for (int s = 64; s > 0; s >>= 1) {
        if (b < s) red[b] += red[b + s];
        __syncthreads();
    }
    if (b == 0) out[0] = red[0] / (float)B_;
}

// --------------------------------------------------------------------------
extern "C" void kernel_launch(void* const* d_in, const int* in_sizes, int n_in,
                              void* d_out, int out_size, void* d_ws, size_t ws_size,
                              hipStream_t stream) {
    const int*   ids     = (const int*)d_in[0];
    const int*   lengths = (const int*)d_in[1];
    const int*   labels  = (const int*)d_in[2];
    const float* emb     = (const float*)d_in[3];
    const float* Wl      = (const float*)d_in[4];
    const float* bl      = (const float*)d_in[5];
    const float* Wd      = (const float*)d_in[6];
    const float* bd      = (const float*)d_in[7];

    char*  ws    = (char*)d_ws;
    size_t szW   = (size_t)KT * NT * 32 * 16 * 2;                // 768000 B each
    size_t offWx = 0;
    size_t offWh = (offWx + szW + 1023) & ~(size_t)1023;
    size_t offH  = (offWh + szW + 1023) & ~(size_t)1023;
    size_t offC  = (offH + (size_t)B_ * KP * 2 + 1023) & ~(size_t)1023;
    size_t offXW = (offC + (size_t)B_ * H_ * 4 + 1023) & ~(size_t)1023;

    _Float16* Wx = (_Float16*)(ws + offWx);
    _Float16* Wh = (_Float16*)(ws + offWh);
    _Float16* h  = (_Float16*)(ws + offH);
    float*    c  = (float*)(ws + offC);
    _Float16* xw = (_Float16*)(ws + offXW);   // [T][B][4H] f16, ~157 MB

    wswizzle<<<(KT * NT * 32 * 16 + 255) / 256, 256, 0, stream>>>(Wl, Wx, Wh);
    embed_gemm<<<(B_ * T_) / 16, 128, 0, stream>>>(ids, emb, Wx, bl, xw);
    lstm_persistent<<<B_ / 16, 256, 0, stream>>>(Wh, xw, lengths, h, c);
    final_dense<<<1, 128, 0, stream>>>(h, Wd, bd, labels, (float*)d_out);
}
